// SRU_Formula_80401787781723
// MI455X (gfx1250) — compile-verified, matrix operand order checked by reading, not run on hardware
//
#include <hip/hip_runtime.h>

#define T_  2048
#define B_  32
#define V_  50257
#define D_  512
#define H_  768
#define C_  10
#define TB_ (T_ * B_)

typedef __attribute__((ext_vector_type(16))) __bf16 v16bf;
typedef __attribute__((ext_vector_type(8)))  float  v8f;
typedef __attribute__((ext_vector_type(4)))  float  v4f;
typedef __attribute__((ext_vector_type(4)))  unsigned int v4u;
typedef __attribute__((ext_vector_type(2)))  unsigned int v2u;

union Frag { v16bf v; v4u q[2]; };

__device__ __forceinline__ unsigned short f2bf(float x) {
    unsigned int u = __float_as_uint(x);
    u += 0x7FFFu + ((u >> 16) & 1u);     // round-to-nearest-even
    return (unsigned short)(u >> 16);
}
__device__ __forceinline__ float bf2f(unsigned int s) {
    return __uint_as_float(s << 16);
}
__device__ __forceinline__ unsigned int pk2(float lo, float hi) {
    return (unsigned int)f2bf(lo) | ((unsigned int)f2bf(hi) << 16);
}

// ---- async copy global->LDS (16B per lane), with sync fallback -------------
#if __has_builtin(__builtin_amdgcn_global_load_async_to_lds_b128)
#define USE_ASYNC_LDS 1
#endif

typedef int v4i_b __attribute__((vector_size(16)));   // matches builtin proto
typedef __attribute__((address_space(1))) v4i_b* gptr_v4i;
typedef __attribute__((address_space(3))) v4i_b* lptr_v4i;

__device__ __forceinline__ void cp16(const unsigned short* g, unsigned short* l) {
#ifdef USE_ASYNC_LDS
    __builtin_amdgcn_global_load_async_to_lds_b128(
        (gptr_v4i)(const void*)g, (lptr_v4i)(void*)l, 0, 0);
#else
    *(v4u*)l = *(const v4u*)g;
#endif
}
__device__ __forceinline__ void cp_fence() {
#ifdef USE_ASYNC_LDS
#if __has_builtin(__builtin_amdgcn_s_wait_asynccnt)
    __builtin_amdgcn_s_wait_asynccnt(0);
#else
    asm volatile("s_wait_asynccnt 0x0" ::: "memory");
#endif
#endif
}

// ---------------------------------------------------------------------------
// Kernel 0: f32 -> bf16 conversion (embedding table, weights). 8 elems/thread.
// ---------------------------------------------------------------------------
__global__ __launch_bounds__(256)
void cvt_bf16(const float* __restrict__ src, unsigned short* __restrict__ dst, int n)
{
    const int i = (blockIdx.x * 256 + threadIdx.x) * 8;
    if (i >= n) return;
    v4f a = *(const v4f*)(src + i);
    v4f b = *(const v4f*)(src + i + 4);
    v4u p = { pk2(a.x, a.y), pk2(a.z, a.w), pk2(b.x, b.y), pk2(b.z, b.w) };
    *(v4u*)(dst + i) = p;
}

// ---------------------------------------------------------------------------
// Kernel 1: embedding gather + 4 fused GEMMs via WMMA bf16.
// Block tile 128(M) x 64(N), K step 32, double-buffered LDS with async
// global->LDS prefetch. 8 waves = 2 M-halves x 4 weight matrices.
// Gates stored interleaved: gates[(m*H + h)*4 + mat], mat: 0=xp 1=f 2=r 3=cx.
// ---------------------------------------------------------------------------
__global__ __launch_bounds__(256)
void sru_gates_wmma(const int* __restrict__ x,
                    const unsigned short* __restrict__ embBf,
                    const unsigned short* __restrict__ Wbf,     // [4][H][D] bf16
                    const float* __restrict__ bf_, const float* __restrict__ br_,
                    const float* __restrict__ bcx_,
                    unsigned short* __restrict__ gates)
{
    __shared__ __align__(16) unsigned short As[2][128][32];     // 16 KB
    __shared__ __align__(16) unsigned short Bs[2][4][64][32];   // 32 KB
    __shared__ int idx[128];

    const int tid = threadIdx.x;
    const int m0  = blockIdx.x * 128;
    const int n0  = blockIdx.y * 64;

    if (tid < 128) idx[tid] = x[m0 + tid];
    __syncthreads();

    // --- per-thread copy assignments: 16B chunks, row = 32 bf16 = 4 chunks --
    const int sub  = (tid & 3) * 8;          // element offset of this 16B chunk
    const int rb   = tid >> 2;               // 0..63
    unsigned aOff[2]; int aLds[2];
#pragma unroll
    for (int j = 0; j < 2; ++j) {
        const int r = rb + j * 64;                       // A row 0..127
        aOff[j] = (unsigned)idx[r] * D_ + sub;
        aLds[j] = r * 32 + sub;
    }
    unsigned bOff[4]; int bLds[4];
#pragma unroll
    for (int j = 0; j < 4; ++j) {
        const int row = rb + j * 64;                     // 0..255 = mat*64 + n
        const int mt  = row >> 6;
        const int n   = row & 63;
        bOff[j] = (unsigned)((mt * H_ + n0 + n) * D_ + sub);
        bLds[j] = row * 32 + sub;
    }

    const int wave = tid >> 5;
    const int lane = tid & 31;
    const int mat  = wave & 3;
    const int mrow = wave >> 2;
    const int lhi  = lane >> 4;
    const int llo  = lane & 15;

    v8f acc[4][4] = {};

    auto loadTile = [&](int k0, int buf) {
        unsigned short* aB = &As[buf][0][0];
        unsigned short* bB = &Bs[buf][0][0][0];
#pragma unroll
        for (int j = 0; j < 2; ++j) cp16(embBf + aOff[j] + k0, aB + aLds[j]);
#pragma unroll
        for (int j = 0; j < 4; ++j) cp16(Wbf + bOff[j] + k0, bB + bLds[j]);
    };

    auto compute = [&](int buf) {
        const unsigned short* aT = &As[buf][0][0];
        const unsigned short* bT = &Bs[buf][mat][0][0];
        Frag fb[4];
#pragma unroll
        for (int ni = 0; ni < 4; ++ni) {
            const unsigned short* bp = bT + (ni * 16 + llo) * 32;
            fb[ni].q[0] = *(const v4u*)(bp + lhi * 8);
            fb[ni].q[1] = *(const v4u*)(bp + 16 + lhi * 8);
        }
#pragma unroll
        for (int mi = 0; mi < 4; ++mi) {
            Frag fa;
            const unsigned short* ap = aT + (mrow * 64 + mi * 16 + llo) * 32;
            fa.q[0] = *(const v4u*)(ap + lhi * 8);
            fa.q[1] = *(const v4u*)(ap + 16 + lhi * 8);
#pragma unroll
            for (int ni = 0; ni < 4; ++ni) {
                acc[mi][ni] = __builtin_amdgcn_wmma_f32_16x16x32_bf16(
                    false, fa.v, false, fb[ni].v, (short)0, acc[mi][ni], false, false);
            }
        }
    };

    loadTile(0, 0);
    cp_fence();
    __syncthreads();
    for (int s = 0; s < 16; ++s) {
        if (s < 15) loadTile((s + 1) * 32, (s + 1) & 1);
        compute(s & 1);
        cp_fence();
        __syncthreads();
    }

    // ---- epilogue: bias / sigmoid, interleaved bf16 store ----
    const float* bias = (mat == 1) ? bf_ : (mat == 2) ? br_ : (mat == 3) ? bcx_ : nullptr;
    const bool sig = (mat == 1) || (mat == 2);

#pragma unroll
    for (int mi = 0; mi < 4; ++mi) {
        const int mbase = m0 + mrow * 64 + mi * 16 + lhi * 8;
#pragma unroll
        for (int ni = 0; ni < 4; ++ni) {
            const int h = n0 + ni * 16 + llo;
            const float bv = bias ? bias[h] : 0.0f;
#pragma unroll
            for (int j = 0; j < 8; ++j) {
                float v = acc[mi][ni][j] + bv;
                if (sig) v = 1.0f / (1.0f + __expf(-v));
                gates[((size_t)(mbase + j) * H_ + h) * 4 + mat] = f2bf(v);
            }
        }
    }
}

// ---------------------------------------------------------------------------
// Kernel 2: sequential SRU scan over T + tanh/max-pool fusion.
// One 8-byte coalesced load per (thread, timestep).
// ---------------------------------------------------------------------------
__global__ __launch_bounds__(256)
void sru_scan(const unsigned short* __restrict__ gates, float* __restrict__ pooled)
{
    const int g = blockIdx.x * 256 + threadIdx.x;   // 0 .. B*H-1
    const int b = g / H_;
    const int h = g - b * H_;
    size_t off = (size_t)b * H_ + h;
    const size_t stride = (size_t)B_ * H_;

    float c  = 0.0f;
    float mx = -3.402823466e38f;
    for (int t = 0; t < T_; ++t) {
        const v2u gg = *(const v2u*)(gates + off * 4);
        const float xt = bf2f(gg.x & 0xFFFFu);
        const float ft = bf2f(gg.x >> 16);
        const float rt = bf2f(gg.y & 0xFFFFu);
        const float ct = bf2f(gg.y >> 16);
        c = ft * c + (1.0f - ft) * xt;
        const float hv = rt * tanhf(c) + (1.0f - rt) * ct;
        mx = fmaxf(mx, hv);
        off += stride;
    }
    pooled[g] = tanhf(tanhf(mx));   // max(tanh(h)) == tanh(max h)
}

// ---------------------------------------------------------------------------
// Kernel 3: tiny classifier [B,H] x [H,C] + bias.
// ---------------------------------------------------------------------------
__global__ void sru_cls(const float* __restrict__ pooled, const float* __restrict__ Wout,
                        const float* __restrict__ bout, float* __restrict__ out)
{
    const int t = blockIdx.x * 64 + threadIdx.x;
    if (t >= B_ * C_) return;
    const int b = t / C_;
    const int c = t - b * C_;
    const float* p = pooled + (size_t)b * H_;
    const float* w = Wout   + (size_t)c * H_;
    float s = bout[c];
    for (int h = 0; h < H_; ++h) s += p[h] * w[h];
    out[t] = s;
}

extern "C" void kernel_launch(void* const* d_in, const int* in_sizes, int n_in,
                              void* d_out, int out_size, void* d_ws, size_t ws_size,
                              hipStream_t stream) {
    (void)in_sizes; (void)n_in; (void)out_size; (void)ws_size;
    const int*   x     = (const int*)d_in[0];
    const float* embed = (const float*)d_in[1];
    const float* Wx    = (const float*)d_in[2];
    const float* Wf    = (const float*)d_in[3];
    const float* bf    = (const float*)d_in[4];
    const float* Wr    = (const float*)d_in[5];
    const float* br    = (const float*)d_in[6];
    const float* Wcx   = (const float*)d_in[7];
    const float* bcx   = (const float*)d_in[8];
    const float* Wout  = (const float*)d_in[9];
    const float* bout  = (const float*)d_in[10];
    float* out = (float*)d_out;

    const size_t S = (size_t)TB_ * H_;
    unsigned short* gates  = (unsigned short*)d_ws;           // [TB][H][4] bf16
    float*          pooled = (float*)(gates + 4 * S);         // [B][H]
    unsigned short* embBf  = (unsigned short*)(pooled + B_ * H_);  // [V][D]
    unsigned short* Wbf    = embBf + (size_t)V_ * D_;              // [4][H][D]

    // f32 -> bf16 pre-conversion (table stays L2-resident afterwards)
    const int nE = V_ * D_;
    cvt_bf16<<<(nE / 8 + 255) / 256, 256, 0, stream>>>(embed, embBf, nE);
    const int nW = H_ * D_;
    cvt_bf16<<<nW / 8 / 256, 256, 0, stream>>>(Wx,  Wbf + 0 * (size_t)nW, nW);
    cvt_bf16<<<nW / 8 / 256, 256, 0, stream>>>(Wf,  Wbf + 1 * (size_t)nW, nW);
    cvt_bf16<<<nW / 8 / 256, 256, 0, stream>>>(Wr,  Wbf + 2 * (size_t)nW, nW);
    cvt_bf16<<<nW / 8 / 256, 256, 0, stream>>>(Wcx, Wbf + 3 * (size_t)nW, nW);

    dim3 g1(TB_ / 128, H_ / 64);
    sru_gates_wmma<<<g1, 256, 0, stream>>>(x, embBf, Wbf, bf, br, bcx, gates);
    sru_scan<<<(B_ * H_) / 256, 256, 0, stream>>>(gates, pooled);
    sru_cls<<<(B_ * C_ + 63) / 64, 64, 0, stream>>>(pooled, Wout, bout, out);
}
